// CLIPCrossProductClassifier_10788957848066
// MI455X (gfx1250) — matrix operand order repaired and checked by reading, not compiled, not run on hardware
//

#include <hip/hip_runtime.h>
#include <hip/hip_bf16.h>

// ---------------------------------------------------------------------------
// CLIP cross-product bilinear classifier for MI455X (gfx1250).
//
// h[b,h'] = sum_{i,j} img_n[b,i] * txt_n[b,j] * W1[(i*D+j), h']   (+b1, relu)
// logits  = h @ W2 + b2
//
// Treated as GEMM  h = A @ W1  with generated A[b, i*D+j] = img[b,i]*txt[b,j].
// A-fragments are built on the fly: txt f16 fragment * per-lane img scalar
// (v_pk_mul_f16), W1 is streamed fp32 -> v_cvt_pk_rtz_f16_f32 -> f16 B-frags,
// accumulated with v_wmma_f32_16x16x32_f16 (f32 accumulators).
//
// Roofline: W1 = 512 MB read once = ~22us @ 23.3 TB/s; 137 GFLOP in f16 WMMA
// lands at the same order -> balanced. i-partitioned grid guarantees W1 is
// read exactly once; intra-WGP 8-wave B-frag redundancy is served by WGP$.
// ---------------------------------------------------------------------------

typedef __attribute__((ext_vector_type(16))) _Float16 v16h;
typedef __attribute__((ext_vector_type(2)))  _Float16 v2h;
typedef __attribute__((ext_vector_type(8)))  float    v8f;

#define Bn 512
#define Dn 512
#define Hn 512
#define ICH 16   // i's per chunk -> Dn/ICH = 32 i-chunks in grid.y

union HalfPack { v16h v; v2h p[8]; };

// --- Kernel 1: L2-normalize rows; write imgT (f32, [D][B]) and txt16 (f16,
// [B][D]); seed hpart with b1. One block (256 thr) per row b. ---------------
__global__ void __launch_bounds__(256)
prep_kernel(const float* __restrict__ img, const float* __restrict__ txt,
            const float* __restrict__ b1,
            float* __restrict__ imgT, _Float16* __restrict__ txt16,
            float* __restrict__ hpart)
{
    const int r = blockIdx.x;
    const int t = threadIdx.x;
    __shared__ float s1[256];
    __shared__ float s2[256];

    const float i0 = img[r * Dn + t];
    const float i1 = img[r * Dn + t + 256];
    const float t0 = txt[r * Dn + t];
    const float t1 = txt[r * Dn + t + 256];
    s1[t] = i0 * i0 + i1 * i1;
    s2[t] = t0 * t0 + t1 * t1;
    __syncthreads();
    for (int off = 128; off > 0; off >>= 1) {
        if (t < off) { s1[t] += s1[t + off]; s2[t] += s2[t + off]; }
        __syncthreads();
    }
    const float invI = 1.0f / fmaxf(sqrtf(s1[0]), 1e-12f);
    const float invT = 1.0f / fmaxf(sqrtf(s2[0]), 1e-12f);

    imgT[t * Bn + r]         = i0 * invI;         // transposed for coalesced
    imgT[(t + 256) * Bn + r] = i1 * invI;         // per-i scalar loads
    txt16[r * Dn + t]        = (_Float16)(t0 * invT);
    txt16[r * Dn + t + 256]  = (_Float16)(t1 * invT);
    hpart[r * Hn + t]        = b1[t];
    hpart[r * Hn + t + 256]  = b1[t + 256];
}

// --- Kernel 2: bilinear GEMM with WMMA f16. ---------------------------------
// Block = 256 thr = 8 waves; wave w covers b-rows [w*64, w*64+64) (all 512 b
// per WG) x 64 h-cols. Wave tile = 4 M-tiles x 4 N-tiles of 16x16.
// grid.x = Hn/64 = 8 h-tiles; grid.y = Dn/ICH = 32 i-chunks.
__global__ void __launch_bounds__(256)
bilinear_wmma(const float* __restrict__ W1, const float* __restrict__ imgT,
              const _Float16* __restrict__ txt16, float* __restrict__ hpart)
{
    const int lane = threadIdx.x & 31;
    const int wave = threadIdx.x >> 5;    // 0..7 (wave32)
    const int half = lane >> 4;           // 0/1
    const int ln   = lane & 15;
    const int h0   = blockIdx.x * 64;
    const int i0   = blockIdx.y * ICH;
    const int bw   = wave * 64;           // this wave's b base

    v8f acc[4][4] = {};

    for (int jc = 0; jc < Dn / 32; ++jc) {
        const int j0 = jc * 32;

        // txt A-fragments (16x32 f16 layout: M=ln; VGPR v pairs:
        // K = 2*(v%4) + 16*(v/4) + 8*half, +1). Constant over the i-loop.
        v16h tf[4];
#pragma unroll
        for (int mt = 0; mt < 4; ++mt) {
            const int row = bw + mt * 16 + ln;
            const _Float16* tb = txt16 + row * Dn + j0 + half * 8;
            HalfPack u;
#pragma unroll
            for (int v = 0; v < 4; ++v)
                u.p[v] = *(const v2h*)(tb + 2 * v);
#pragma unroll
            for (int v = 4; v < 8; ++v)
                u.p[v] = *(const v2h*)(tb + 16 + 2 * (v - 4));
            tf[mt] = u.v;
        }

        for (int ii = 0; ii < ICH; ++ii) {
            const int i = i0 + ii;

            // A = img[b,i] * txt[b,j]  (per-lane scalar * fragment)
            v16h a[4];
#pragma unroll
            for (int mt = 0; mt < 4; ++mt) {
                const float s = imgT[i * Bn + bw + mt * 16 + ln];
                a[mt] = tf[mt] * (_Float16)s;   // 8x v_pk_mul_f16
            }

            // B fragments: W1[(i*D + j), h], j = j0 + 8*half + e + 8*(e>=8),
            // N = ln. fp32 loads -> v_cvt_pk_rtz_f16_f32.
            const float* wbase =
                W1 + (size_t)((unsigned)(i * Dn + j0 + half * 8) * (unsigned)Hn)
                   + h0 + ln;
#pragma unroll
            for (int nt = 0; nt < 4; ++nt) {
                const float* wb = wbase + nt * 16;
                HalfPack bu;
#pragma unroll
                for (int v = 0; v < 8; ++v) {
                    const float f0 = wb[(2 * v + 8 * (v >> 2))     * Hn];
                    const float f1 = wb[(2 * v + 8 * (v >> 2) + 1) * Hn];
                    bu.p[v] = __builtin_bit_cast(
                        v2h, __builtin_amdgcn_cvt_pkrtz(f0, f1));
                }
#pragma unroll
                for (int mt = 0; mt < 4; ++mt)
                    acc[mt][nt] = __builtin_amdgcn_wmma_f32_16x16x32_f16(
                        false, a[mt], false, bu.v, (short)0, acc[mt][nt],
                        false, false);
            }
        }
    }

    // C/D layout: VGPR v -> M = v + 8*half, N = ln. Accumulate partials.
#pragma unroll
    for (int mt = 0; mt < 4; ++mt)
#pragma unroll
        for (int nt = 0; nt < 4; ++nt) {
            const int hcol = h0 + nt * 16 + ln;
#pragma unroll
            for (int v = 0; v < 8; ++v) {
                const int brow = bw + mt * 16 + half * 8 + v;
                unsafeAtomicAdd(&hpart[brow * Hn + hcol], acc[mt][nt][v]);
            }
        }
}

// --- Kernel 3: logits[b] = relu(h[b,:]) @ W2 + b2. One block per row. ------
__global__ void __launch_bounds__(256)
head_kernel(const float* __restrict__ hpart, const float* __restrict__ W2,
            const float* __restrict__ b2, float* __restrict__ out)
{
    const int r = blockIdx.x;
    const int t = threadIdx.x;
    __shared__ float s[256];
    const float v0 = hpart[r * Hn + t];
    const float v1 = hpart[r * Hn + t + 256];
    s[t] = fmaxf(v0, 0.0f) * W2[t] + fmaxf(v1, 0.0f) * W2[t + 256];
    __syncthreads();
    for (int off = 128; off > 0; off >>= 1) {
        if (t < off) s[t] += s[t + off];
        __syncthreads();
    }
    if (t == 0) out[r] = s[0] + b2[0];
}

extern "C" void kernel_launch(void* const* d_in, const int* in_sizes, int n_in,
                              void* d_out, int out_size, void* d_ws, size_t ws_size,
                              hipStream_t stream) {
    const float* img = (const float*)d_in[0];   // [B, D] f32
    const float* txt = (const float*)d_in[1];   // [B, D] f32
    const float* W1  = (const float*)d_in[2];   // [D*D, H] f32
    const float* b1  = (const float*)d_in[3];   // [H] f32
    const float* W2  = (const float*)d_in[4];   // [H, 1] f32
    const float* b2  = (const float*)d_in[5];   // [1] f32
    float* out = (float*)d_out;                 // [B, 1] f32

    // Workspace layout (2.5 MB total):
    char* ws = (char*)d_ws;
    float*    imgT  = (float*)ws;                               // 1 MB  [D][B]
    _Float16* txt16 = (_Float16*)(ws + (1u << 20));             // 512KB [B][D]
    float*    hpart = (float*)(ws + (1u << 20) + (1u << 19));   // 1 MB  [B][H]

    prep_kernel<<<Bn, 256, 0, stream>>>(img, txt, b1, imgT, txt16, hpart);

    dim3 grid(Hn / 64, Dn / ICH);   // 8 x 32 = 256 workgroups
    bilinear_wmma<<<grid, 256, 0, stream>>>(W1, imgT, txt16, hpart);

    head_kernel<<<Bn, 256, 0, stream>>>(hpart, W2, b2, out);
}
